// MultiHeadAttention_53017076301867
// MI455X (gfx1250) — compile-verified, matrix operand order checked by reading, not run on hardware
//
#include <hip/hip_runtime.h>

typedef __bf16 bf16;
typedef __attribute__((ext_vector_type(16))) __bf16 v16bf;
typedef __attribute__((ext_vector_type(8)))  __bf16 v8bf;
typedef __attribute__((ext_vector_type(8)))  float  v8f;
typedef __attribute__((ext_vector_type(4)))  unsigned int u32x4;
typedef __attribute__((ext_vector_type(8)))  int i32x8;
typedef __attribute__((ext_vector_type(4)))  int i32x4;

#define S_   2048
#define E_   1024
#define H_   16
#define DH_  64
#define WIN_ 256
#define CH_  8
#define SCALE_ 0.125f            /* 1/sqrt(64) */
#define NEGM_  (-1.25e8f)        /* -1e9 * scale, matches reference mask-then-scale */

__device__ __forceinline__ bf16 f2bf(float f) {
  unsigned u = __builtin_bit_cast(unsigned, f);
  unsigned r = u + 0x7FFFu + ((u >> 16) & 1u);
  unsigned short h = (unsigned short)(r >> 16);
  return __builtin_bit_cast(bf16, h);
}
__device__ __forceinline__ float bf2f(bf16 b) {
  unsigned short h = __builtin_bit_cast(unsigned short, b);
  unsigned u = ((unsigned)h) << 16;
  return __builtin_bit_cast(float, u);
}
__device__ __forceinline__ v8f wmma_bf16(v16bf a, v16bf b, v8f c) {
  return __builtin_amdgcn_wmma_f32_16x16x32_bf16(false, a, false, b, (short)0, c, false, false);
}
// Loads one 16-bit A/B fragment half-pair: 8 elems at p, 8 elems at p+16.
__device__ __forceinline__ v16bf ld_frag(const bf16* p) {
  v16bf r;
  *(v8bf*)&r       = *(const v8bf*)p;
  *((v8bf*)&r + 1) = *(const v8bf*)(p + 16);
  return r;
}

// ---- Tensor Data Mover: 2-D bf16 tile global -> LDS (CDNA5 ISA) ----
// D# group0: count=1, is_restore=0, gather off; lds_addr; global_addr[56:0]; type=2.
// D# group1: wg_mask=0, data_size=1 (2B), no barrier/iterate/pad; tensor dims == tile
// dims (no OOB path); dim0_stride in elements. This toolchain exposes the 6-arg
// builtin (extra descriptor group), zero-filled since the 2-D form doesn't use it.
__device__ __forceinline__ void tdm_load_2d(unsigned lds_off, unsigned long long gaddr,
                                            unsigned tile_k, unsigned tile_rows,
                                            unsigned long long stride_elems) {
  u32x4 g0;
  g0[0] = 1u;                                             // count=1
  g0[1] = lds_off;                                        // lds_addr (bytes)
  g0[2] = (unsigned)(gaddr & 0xFFFFFFFFull);              // global_addr[31:0]
  g0[3] = (unsigned)((gaddr >> 32) & 0x1FFFFFFull) | (2u << 30);  // addr[56:32] | type=2
  i32x8 g1;
  g1[0] = (int)(1u << 16);                                // data_size=1 -> 2 bytes
  g1[1] = (int)((tile_k & 0xFFFFu) << 16);                // tensor_dim0[15:0] (== tile_k)
  g1[2] = (int)((tile_k >> 16) | ((tile_rows & 0xFFFFu) << 16));  // dim0 hi | dim1 lo
  g1[3] = (int)((tile_rows >> 16) | (tile_k << 16));      // dim1 hi | tile_dim0
  g1[4] = (int)tile_rows;                                 // tile_dim1 | tile_dim2=0
  g1[5] = (int)(stride_elems & 0xFFFFFFFFull);            // dim0_stride[31:0]
  g1[6] = (int)((stride_elems >> 32) & 0xFFFFull);        // stride[47:32] | dim1_stride lo=0
  g1[7] = 0;
  i32x4 gz4 = {0, 0, 0, 0};
  i32x8 gz8 = {0, 0, 0, 0, 0, 0, 0, 0};
  __builtin_amdgcn_tensor_load_to_lds(g0, g1, gz4, gz4, gz8, 0);
}

// ---------------- conversion kernels ----------------
__global__ void k_cvt_bf16(const float* __restrict__ src, bf16* __restrict__ dst, int n) {
  int i = blockIdx.x * blockDim.x + threadIdx.x;
  if (i < n) dst[i] = f2bf(src[i]);
}
// W [K=E, N=E] f32 row-major -> Wt [N, K] bf16 row-major (K contiguous for B fragments)
__global__ void k_cvt_wT(const float* __restrict__ W, bf16* __restrict__ Wt) {
  int i = blockIdx.x * blockDim.x + threadIdx.x;
  int k = i >> 10, n = i & (E_ - 1);
  Wt[(size_t)n * E_ + k] = f2bf(W[(size_t)k * E_ + n]);
}
// Vt[h][d][s] = Yvg[s][h*64+d]   (keys contiguous for PV B fragments)
__global__ void k_vt_g(const bf16* __restrict__ Yvg, bf16* __restrict__ Vt) {
  int i = blockIdx.x * blockDim.x + threadIdx.x;   // i = ((h*64+d)*2048 + s)
  int s = i & (S_ - 1);
  int hd = i >> 11;                                 // h*64+d
  Vt[i] = Yvg[(size_t)s * E_ + hd];
}

// ------- batched projection GEMM: Y[z] = X @ W[z], bf16 out, TDM-staged A tile -------
// Block 256 = 8 waves, workgroup tile 64M x 128N, wave tile 32M x 32N.
__global__ __launch_bounds__(256) void k_gemm_bf16(const bf16* __restrict__ A,
                                                   const bf16* __restrict__ WtBase,
                                                   bf16* __restrict__ YBase) {
  __shared__ __align__(16) bf16 sA[64 * 128];
  const bf16* Bt = WtBase + (size_t)blockIdx.z * E_ * E_;
  bf16* C = YBase + (size_t)blockIdx.z * S_ * E_;
  const int lane = threadIdx.x & 31, wid = threadIdx.x >> 5;
  const int l15 = lane & 15, hi = lane >> 4, kd = hi * 8;
  const int rowBase = blockIdx.y * 64;
  const int colBase = blockIdx.x * 128 + (wid >> 1) * 32;
  const int wm = (wid & 1) * 32;
  const bf16* b0r = Bt + (size_t)(colBase + l15) * E_;
  const bf16* b1r = Bt + (size_t)(colBase + 16 + l15) * E_;
  const unsigned ldsOff = (unsigned)(unsigned long long)(void*)sA;
  v8f acc00 = {}, acc01 = {}, acc10 = {}, acc11 = {};
  for (int k0 = 0; k0 < E_; k0 += 128) {
    __syncthreads();                       // previous tile fully consumed
    if (wid == 0) {
      unsigned long long ga = (unsigned long long)(A + (size_t)rowBase * E_ + k0);
      tdm_load_2d(ldsOff, ga, 128u, 64u, (unsigned long long)E_);
      __builtin_amdgcn_s_wait_tensorcnt(0);
    }
    __syncthreads();                       // tile visible to all waves
    if (k0 + 128 < E_) {                   // warm L2 for next B chunk
      __builtin_prefetch(b0r + k0 + 128, 0, 1);
      __builtin_prefetch(b1r + k0 + 128, 0, 1);
    }
#pragma unroll
    for (int kk = 0; kk < 128; kk += 32) {
      v16bf a0 = ld_frag(sA + (wm + l15) * 128 + kk + kd);
      v16bf a1 = ld_frag(sA + (wm + 16 + l15) * 128 + kk + kd);
      v16bf b0 = ld_frag(b0r + k0 + kk + kd);
      v16bf b1 = ld_frag(b1r + k0 + kk + kd);
      acc00 = wmma_bf16(a0, b0, acc00);
      acc01 = wmma_bf16(a0, b1, acc01);
      acc10 = wmma_bf16(a1, b0, acc10);
      acc11 = wmma_bf16(a1, b1, acc11);
    }
  }
#pragma unroll
  for (int r = 0; r < 8; ++r) {
    int ra = rowBase + wm + r + hi * 8;
    int rb = ra + 16;
    C[(size_t)ra * E_ + colBase + l15]      = f2bf(acc00[r]);
    C[(size_t)ra * E_ + colBase + 16 + l15] = f2bf(acc01[r]);
    C[(size_t)rb * E_ + colBase + l15]      = f2bf(acc10[r]);
    C[(size_t)rb * E_ + colBase + 16 + l15] = f2bf(acc11[r]);
  }
}

// ---------------- output GEMM with bias, f32 out (same TDM structure) ----------------
__global__ __launch_bounds__(256) void k_gemm_out(const bf16* __restrict__ A,
                                                  const bf16* __restrict__ Bt,
                                                  const float* __restrict__ bias,
                                                  float* __restrict__ C) {
  __shared__ __align__(16) bf16 sA[64 * 128];
  const int lane = threadIdx.x & 31, wid = threadIdx.x >> 5;
  const int l15 = lane & 15, hi = lane >> 4, kd = hi * 8;
  const int rowBase = blockIdx.y * 64;
  const int colBase = blockIdx.x * 128 + (wid >> 1) * 32;
  const int wm = (wid & 1) * 32;
  const bf16* b0r = Bt + (size_t)(colBase + l15) * E_;
  const bf16* b1r = Bt + (size_t)(colBase + 16 + l15) * E_;
  const unsigned ldsOff = (unsigned)(unsigned long long)(void*)sA;
  v8f acc00 = {}, acc01 = {}, acc10 = {}, acc11 = {};
  for (int k0 = 0; k0 < E_; k0 += 128) {
    __syncthreads();
    if (wid == 0) {
      unsigned long long ga = (unsigned long long)(A + (size_t)rowBase * E_ + k0);
      tdm_load_2d(ldsOff, ga, 128u, 64u, (unsigned long long)E_);
      __builtin_amdgcn_s_wait_tensorcnt(0);
    }
    __syncthreads();
    if (k0 + 128 < E_) {
      __builtin_prefetch(b0r + k0 + 128, 0, 1);
      __builtin_prefetch(b1r + k0 + 128, 0, 1);
    }
#pragma unroll
    for (int kk = 0; kk < 128; kk += 32) {
      v16bf a0 = ld_frag(sA + (wm + l15) * 128 + kk + kd);
      v16bf a1 = ld_frag(sA + (wm + 16 + l15) * 128 + kk + kd);
      v16bf b0 = ld_frag(b0r + k0 + kk + kd);
      v16bf b1 = ld_frag(b1r + k0 + kk + kd);
      acc00 = wmma_bf16(a0, b0, acc00);
      acc01 = wmma_bf16(a0, b1, acc01);
      acc10 = wmma_bf16(a1, b0, acc10);
      acc11 = wmma_bf16(a1, b1, acc11);
    }
  }
  float bia0 = bias[colBase + l15], bia1 = bias[colBase + 16 + l15];
#pragma unroll
  for (int r = 0; r < 8; ++r) {
    int ra = rowBase + wm + r + hi * 8;
    int rb = ra + 16;
    C[(size_t)ra * E_ + colBase + l15]      = acc00[r] + bia0;
    C[(size_t)ra * E_ + colBase + 16 + l15] = acc01[r] + bia1;
    C[(size_t)rb * E_ + colBase + l15]      = acc10[r] + bia0;
    C[(size_t)rb * E_ + colBase + 16 + l15] = acc11[r] + bia1;
  }
}

// ---------------- global branch: causal flash attention (WMMA) ----------------
// grid (S/128, H), block 256: wave w owns 16 query rows q0 = blk*128 + w*16.
__global__ __launch_bounds__(256) void k_flash_global(const bf16* __restrict__ Qg,
                                                      const bf16* __restrict__ Kg,
                                                      const bf16* __restrict__ Vt,
                                                      float* __restrict__ gctx) {
  __shared__ __align__(16) bf16 pbuf[8][16][32];
  const int lane = threadIdx.x & 31, wid = threadIdx.x >> 5;
  const int l15 = lane & 15, hi = lane >> 4;
  const int h = blockIdx.y;
  const int q0 = blockIdx.x * 128 + wid * 16;

  const bf16* qrow = Qg + (size_t)(q0 + l15) * E_ + h * DH_;
  const int kd = hi * 8;
  v16bf aq0 = ld_frag(qrow + kd);        // dh 0..31 chunk
  v16bf aq1 = ld_frag(qrow + 32 + kd);   // dh 32..63 chunk

  float m[8], lsum[8];
#pragma unroll
  for (int r = 0; r < 8; ++r) { m[r] = -1e30f; lsum[r] = 0.f; }
  v8f o0 = {}, o1 = {}, o2 = {}, o3 = {};

  for (int kb = 0; kb < q0 + 16; kb += 32) {
    // ---- S tile = Q K^T over dh=64 (2 wmma per 16-key subtile) ----
    const bf16* k0p = Kg + (size_t)(kb + l15) * E_ + h * DH_;
    const bf16* k1p = Kg + (size_t)(kb + 16 + l15) * E_ + h * DH_;
    v8f s0 = {}, s1 = {};
    s0 = wmma_bf16(aq0, ld_frag(k0p + kd), s0);
    s0 = wmma_bf16(aq1, ld_frag(k0p + 32 + kd), s0);
    s1 = wmma_bf16(aq0, ld_frag(k1p + kd), s1);
    s1 = wmma_bf16(aq1, ld_frag(k1p + 32 + kd), s1);

    const int key0 = kb + l15, key1 = kb + 16 + l15;
#pragma unroll
    for (int r = 0; r < 8; ++r) {
      int q = q0 + r + hi * 8;
      s0[r] = (key0 <= q) ? s0[r] * SCALE_ : NEGM_;
      s1[r] = (key1 <= q) ? s1[r] * SCALE_ : NEGM_;
      // row max across the 16 lanes holding this row
      float t = fmaxf(s0[r], s1[r]);
      t = fmaxf(t, __shfl_xor(t, 1, 32));
      t = fmaxf(t, __shfl_xor(t, 2, 32));
      t = fmaxf(t, __shfl_xor(t, 4, 32));
      t = fmaxf(t, __shfl_xor(t, 8, 32));
      float mn = fmaxf(m[r], t);
      float f = __expf(m[r] - mn);
      float p0 = __expf(s0[r] - mn);
      float p1 = __expf(s1[r] - mn);
      float rs = p0 + p1;
      rs += __shfl_xor(rs, 1, 32);
      rs += __shfl_xor(rs, 2, 32);
      rs += __shfl_xor(rs, 4, 32);
      rs += __shfl_xor(rs, 8, 32);
      lsum[r] = lsum[r] * f + rs;
      m[r] = mn;
      o0[r] *= f; o1[r] *= f; o2[r] *= f; o3[r] *= f;
      // C-layout -> LDS so we can re-read in A-layout for P@V
      pbuf[wid][r + hi * 8][l15]      = f2bf(p0);
      pbuf[wid][r + hi * 8][16 + l15] = f2bf(p1);
    }
    asm volatile("s_wait_dscnt 0" ::: "memory");
    v16bf pa = ld_frag(&pbuf[wid][l15][kd]);   // row=l15, K runs kd / kd+16

    // ---- O += P @ V, V^T supplied key-major from Vt[h][d][s] ----
    const bf16* v0 = Vt + ((size_t)(h * DH_ + 0 * 16 + l15)) * S_ + kb + kd;
    const bf16* v1 = Vt + ((size_t)(h * DH_ + 1 * 16 + l15)) * S_ + kb + kd;
    const bf16* v2 = Vt + ((size_t)(h * DH_ + 2 * 16 + l15)) * S_ + kb + kd;
    const bf16* v3 = Vt + ((size_t)(h * DH_ + 3 * 16 + l15)) * S_ + kb + kd;
    o0 = wmma_bf16(pa, ld_frag(v0), o0);
    o1 = wmma_bf16(pa, ld_frag(v1), o1);
    o2 = wmma_bf16(pa, ld_frag(v2), o2);
    o3 = wmma_bf16(pa, ld_frag(v3), o3);
  }
#pragma unroll
  for (int r = 0; r < 8; ++r) {
    int q = q0 + r + hi * 8;
    float inv = 1.0f / lsum[r];
    float* out = gctx + (size_t)q * E_ + h * DH_;
    out[0 * 16 + l15] = o0[r] * inv;
    out[1 * 16 + l15] = o1[r] * inv;
    out[2 * 16 + l15] = o2[r] * inv;
    out[3 * 16 + l15] = o3[r] * inv;
  }
}

// ---------------- local branch: chunk-summed masked scores (WMMA) ----------------
// sbar[h][q][k] = causal? (scale/8)*sum_c q_c.k_c  :  -1e9*scale
// Window-row tile q in [16t,16t+16) of (h,c) is one memory row
// s = h*128 + c*16 + t reinterpreted as 16x64 (bug-compatible reshape).
__global__ __launch_bounds__(256) void k_local_scores(const bf16* __restrict__ Yql,
                                                      const bf16* __restrict__ Ykl,
                                                      float* __restrict__ sbar) {
  const int lane = threadIdx.x & 31, wid = threadIdx.x >> 5;
  const int l15 = lane & 15, hi = lane >> 4;
  const int kt = blockIdx.x * 8 + wid;   // key tile 0..15
  const int qt = blockIdx.y;             // query tile 0..15
  const int h = blockIdx.z;
  const int kd = hi * 8;
  v8f acc = {};
  for (int c = 0; c < CH_; ++c) {
    const bf16* ar = Yql + (size_t)(h * 128 + c * 16 + qt) * E_ + l15 * DH_;
    const bf16* br = Ykl + (size_t)(h * 128 + c * 16 + kt) * E_ + l15 * DH_;
    acc = wmma_bf16(ld_frag(ar + kd),      ld_frag(br + kd),      acc);
    acc = wmma_bf16(ld_frag(ar + 32 + kd), ld_frag(br + 32 + kd), acc);
  }
  const int key = kt * 16 + l15;
#pragma unroll
  for (int r = 0; r < 8; ++r) {
    int q = qt * 16 + r + hi * 8;
    float v = (key <= q) ? acc[r] * (SCALE_ / (float)CH_) : NEGM_;
    sbar[((size_t)h * WIN_ + q) * WIN_ + key] = v;
  }
}

// Column sums of vloc per head (all rows, and rows >= 256)
__global__ void k_vsum(const bf16* __restrict__ Yvl, float* __restrict__ sumAll,
                       float* __restrict__ sumTail) {
  int h = blockIdx.x, d = threadIdx.x;
  float sa = 0.f, st = 0.f;
  for (int k = 0; k < S_; ++k) {
    float v = bf2f(Yvl[(size_t)(h * 128 + (k >> 4)) * E_ + (k & 15) * DH_ + d]);
    sa += v;
    if (k >= WIN_) st += v;
  }
  sumAll[h * DH_ + d] = sa;
  sumTail[h * DH_ + d] = st;
}

// Analytic padded softmax + context for the local branch
__global__ void k_local_ctx(const float* __restrict__ sbar, const bf16* __restrict__ Yvl,
                            const float* __restrict__ sumAll, const float* __restrict__ sumTail,
                            float* __restrict__ lctx) {
  int q = blockIdx.x, h = blockIdx.y, d = threadIdx.x;
  float out;
  if (q >= WIN_) {
    out = sumAll[h * DH_ + d] * (1.0f / (float)S_);  // uniform row -> mean of all V
  } else {
    float acc = sumTail[h * DH_ + d];                // zero-logit tail keys, weight e^0
    float z = (float)(S_ - WIN_);
    const float* srow = sbar + ((size_t)h * WIN_ + q) * WIN_;
    for (int k = 0; k < WIN_; ++k) {
      float w = __expf(srow[k]);
      acc += w * bf2f(Yvl[(size_t)(h * 128 + (k >> 4)) * E_ + (k & 15) * DH_ + d]);
      z += w;
    }
    out = acc / z;
  }
  lctx[(size_t)q * E_ + h * DH_ + d] = out;
}

// mask-select + bf16 convert for the output projection
__global__ void k_combine(const float* __restrict__ gctx, const float* __restrict__ lctx,
                          const int* __restrict__ gmask, bf16* __restrict__ ctxb) {
  int i = blockIdx.x * blockDim.x + threadIdx.x;
  int q = i >> 10;
  float v = (gmask[q] == 1) ? gctx[i] : lctx[i];
  ctxb[i] = f2bf(v);
}

extern "C" void kernel_launch(void* const* d_in, const int* in_sizes, int n_in,
                              void* d_out, int out_size, void* d_ws, size_t ws_size,
                              hipStream_t stream) {
  (void)in_sizes; (void)n_in; (void)out_size; (void)ws_size;
  const float* x     = (const float*)d_in[0];
  const int*   gmask = (const int*)d_in[1];
  const float* Wsrc[7] = { (const float*)d_in[2], (const float*)d_in[3], (const float*)d_in[4],
                           (const float*)d_in[5], (const float*)d_in[6], (const float*)d_in[7],
                           (const float*)d_in[8] };
  const float* bproj = (const float*)d_in[9];
  float* out = (float*)d_out;

  char* w = (char*)d_ws;
  auto alloc = [&](size_t bytes) { char* p = w; w += (bytes + 255) & ~(size_t)255; return p; };
  const size_t SE = (size_t)S_ * E_;
  bf16*  xb      = (bf16*)alloc(SE * 2);
  bf16*  wT      = (bf16*)alloc(7 * (size_t)E_ * E_ * 2);
  bf16*  y       = (bf16*)alloc(6 * SE * 2);          // ql,kl,vl,qg,kg,vg
  bf16*  vtg     = (bf16*)alloc((size_t)H_ * DH_ * S_ * 2);
  float* sbar    = (float*)alloc((size_t)H_ * WIN_ * WIN_ * 4);
  float* sumAll  = (float*)alloc((size_t)H_ * DH_ * 4);
  float* sumTail = (float*)alloc((size_t)H_ * DH_ * 4);
  float* gctx    = (float*)alloc(SE * 4);
  float* lctx    = (float*)alloc(SE * 4);
  bf16*  ctxb    = (bf16*)alloc(SE * 2);

  // 1) x -> bf16
  k_cvt_bf16<<<dim3(SE / 256), dim3(256), 0, stream>>>(x, xb, (int)SE);
  // 2) weights -> bf16 transposed (K-major)
  for (int i = 0; i < 7; ++i)
    k_cvt_wT<<<dim3((E_ * E_) / 256), dim3(256), 0, stream>>>(Wsrc[i], wT + (size_t)i * E_ * E_);
  // 3) six projections in one batched WMMA GEMM (TDM-staged A tiles)
  k_gemm_bf16<<<dim3(E_ / 128, S_ / 64, 6), dim3(256), 0, stream>>>(xb, wT, y);
  // 4) repack global V -> Vt[h][d][s]
  k_vt_g<<<dim3((H_ * DH_ * S_) / 256), dim3(256), 0, stream>>>(y + 5 * SE, vtg);
  // 5) global causal flash attention
  k_flash_global<<<dim3(S_ / 128, H_), dim3(256), 0, stream>>>(y + 3 * SE, y + 4 * SE, vtg, gctx);
  // 6) local chunk-mean masked scores
  k_local_scores<<<dim3(2, WIN_ / 16, H_), dim3(256), 0, stream>>>(y + 0 * SE, y + 1 * SE, sbar);
  // 7) V column sums
  k_vsum<<<dim3(H_), dim3(DH_), 0, stream>>>(y + 2 * SE, sumAll, sumTail);
  // 8) analytic padded softmax + local context
  k_local_ctx<<<dim3(S_, H_), dim3(DH_), 0, stream>>>(sbar, y + 2 * SE, sumAll, sumTail, lctx);
  // 9) mask select + convert
  k_combine<<<dim3(SE / 256), dim3(256), 0, stream>>>(gctx, lctx, gmask, ctxb);
  // 10) output projection + bias
  k_gemm_out<<<dim3(E_ / 128, S_ / 64), dim3(256), 0, stream>>>(ctxb, wT + 6 * (size_t)E_ * E_, bproj, out);
}
